// DAFusion_38302518346144
// MI455X (gfx1250) — matrix operand order repaired and checked by reading, not compiled, not run on hardware
//
#include <hip/hip_runtime.h>
#include <cstddef>
#include <cstdint>

typedef __attribute__((ext_vector_type(8)))  float  v8f;
typedef __attribute__((ext_vector_type(16))) __bf16 v16bf;
typedef __attribute__((ext_vector_type(8)))  __bf16 v8bf;

#define TM 128
#define TN 64
#define TK 32
#define LDSP (TK + 8)  // bf16 elems; 80B row stride keeps every 16B chunk aligned

enum { EPI_NONE = 0, EPI_BIAS = 1, EPI_RELU = 2, EPI_GELU = 3, EPI_RES = 4 };

// gfx1250 async global->LDS path, compile-guarded so either toolchain builds.
#if defined(__AMDGCN__) && __has_builtin(__builtin_amdgcn_global_load_async_to_lds_b128) && \
    __has_builtin(__builtin_amdgcn_s_wait_asynccnt)
#define USE_ASYNC_LDS 1
#else
#define USE_ASYNC_LDS 0
#endif

// Builtin expects (int4 addrspace(1)*, int4 addrspace(3)*, imm offset, imm cpol):
// clang prints addrspace(1) as "__device__" in HIP diagnostics.
typedef int i32x4_vs __attribute__((vector_size(16)));
typedef __attribute__((address_space(1))) i32x4_vs* gl_i4p;
typedef __attribute__((address_space(3))) i32x4_vs* ls_i4p;

// 16-byte global->LDS copy (8 bf16). Async on gfx1250 when available.
__device__ __forceinline__ void copy16_g2l(__bf16* dst, const __bf16* src) {
#if USE_ASYNC_LDS
  void* s = (void*)src;  // drop const; generic -> as(1)/as(3) via C-style cast
  __builtin_amdgcn_global_load_async_to_lds_b128((gl_i4p)s, (ls_i4p)(void*)dst,
                                                 0, 0);
#else
  *(int4*)dst = *(const int4*)src;
#endif
}
__device__ __forceinline__ void async_wait0() {
#if USE_ASYNC_LDS
  __builtin_amdgcn_s_wait_asynccnt(0);
#endif
}

__device__ __forceinline__ float wave_sum(float v) {
#pragma unroll
  for (int o = 16; o; o >>= 1) v += __shfl_xor(v, o, 32);
  return v;
}
__device__ __forceinline__ float wave_max(float v) {
#pragma unroll
  for (int o = 16; o; o >>= 1) v = fmaxf(v, __shfl_xor(v, o, 32));
  return v;
}

// ---------------------------------------------------------------------------
// Batched GEMM: C = epi(alpha * A @ B (+bias) (+residual))
//  TA/TB in {float,__bf16}: fp32 operands convert during LDS staging; bf16
//  operands stream straight global->LDS (async b128 when BT/contiguous).
//  Double-buffered LDS, register-staged pipeline: next tile's loads issue
//  before the WMMAs of the current tile.
//  z = b*Hdim + h; per-operand element offsets b*s?b + h*s?h.
// ---------------------------------------------------------------------------
template <int EPI, bool BT, typename TA, typename TB, typename TC>
__global__ __launch_bounds__(256) void gemm_wmma_k(
    const TA* __restrict__ Abase, const TB* __restrict__ Bbase,
    TC* __restrict__ Cbase, const float* __restrict__ bias,
    const TC* __restrict__ Rbase, int M, int N, int K, int lda, int ldb,
    int ldc, int ldr, long long sAb, long long sAh, long long sBb,
    long long sBh, long long sCb, long long sCh, long long sRb, long long sRh,
    int Hdim, float alpha) {
  constexpr bool AF32 = __is_same(TA, float);
  constexpr bool BF32 = __is_same(TB, float);

  const int z = blockIdx.z;
  const int bb = z / Hdim, hh = z - bb * Hdim;
  const TA* Aop = Abase + bb * sAb + hh * sAh;
  const TB* Bop = Bbase + bb * sBb + hh * sBh;
  TC* C = Cbase + bb * sCb + hh * sCh;
  const TC* R = Rbase ? (Rbase + bb * sRb + hh * sRh) : nullptr;

  const int tM = blockIdx.y * TM;
  const int tN = blockIdx.x * TN;

  __shared__ __align__(16) __bf16 As[2][TM][LDSP];
  __shared__ __align__(16) __bf16 Bs[2][TN][LDSP];

  const int tid = threadIdx.x;
  const int lane = tid & 31;
  const int wave = tid >> 5;
  const int wm = wave >> 1;
  const int wn = wave & 1;
  const int mi = lane & 15;
  const int kg = lane >> 4;

  float4 aReg[4];   // fp32-A staging
  float bRegF[8];   // fp32-B staging
  __bf16 bRegH[8];  // bf16-B gather staging

  // ---- stage issue: start global reads / async copies for tile k0 -> buf ----
  auto issueA = [&](int k0, int buf) {
    if constexpr (AF32) {
      const int kk = (tid & 7) << 2;
      const int mr = tid >> 3;
#pragma unroll
      for (int p = 0; p < 4; ++p) {
        const int gm = tM + mr + (p << 5);
        float4 f = {0.f, 0.f, 0.f, 0.f};
        if (gm < M) {
          f = *(const float4*)(Aop + (size_t)gm * lda + k0 + kk);
          if (p == 0 && k0 + TK < K)
            __builtin_prefetch(Aop + (size_t)gm * lda + k0 + TK + kk, 0, 1);
        }
        aReg[p] = f;
      }
    } else {
      const int kk = (tid & 3) << 3;
      const int mr = tid >> 2;
#pragma unroll
      for (int p = 0; p < 2; ++p) {
        const int m = mr + (p << 6);
        const int gm = tM + m;
        if (gm < M) copy16_g2l(&As[buf][m][kk], Aop + (size_t)gm * lda + k0 + kk);
      }
    }
  };
  auto commitA = [&](int buf) {
    if constexpr (AF32) {
      const int kk = (tid & 7) << 2;
      const int mr = tid >> 3;
#pragma unroll
      for (int p = 0; p < 4; ++p) {
        const int m = mr + (p << 5);
        As[buf][m][kk + 0] = (__bf16)aReg[p].x;
        As[buf][m][kk + 1] = (__bf16)aReg[p].y;
        As[buf][m][kk + 2] = (__bf16)aReg[p].z;
        As[buf][m][kk + 3] = (__bf16)aReg[p].w;
      }
    }
  };
  auto issueB = [&](int k0, int buf) {
    if constexpr (BF32) {  // fp32 weights [K,N]: gather along N, convert later
      const int n = tid & 63;
      const int kk0 = (tid >> 6) << 3;
      const int gn = tN + n;
#pragma unroll
      for (int j = 0; j < 8; ++j)
        bRegF[j] = (gn < N) ? Bop[(size_t)(k0 + kk0 + j) * ldb + gn] : 0.f;
    } else if constexpr (BT) {  // bf16 [N,K]: contiguous K -> async b128
      const int n = tid >> 2;
      const int kk = (tid & 3) << 3;
      const int gn = tN + n;
      if (gn < N) copy16_g2l(&Bs[buf][n][kk], Bop + (size_t)gn * ldb + k0 + kk);
    } else {  // bf16 [K,N]: gather
      const int n = tid & 63;
      const int kk0 = (tid >> 6) << 3;
      const int gn = tN + n;
#pragma unroll
      for (int j = 0; j < 8; ++j)
        bRegH[j] = (gn < N) ? Bop[(size_t)(k0 + kk0 + j) * ldb + gn] : (__bf16)0.f;
    }
  };
  auto commitB = [&](int buf) {
    if constexpr (BF32) {
      const int n = tid & 63;
      const int kk0 = (tid >> 6) << 3;
#pragma unroll
      for (int j = 0; j < 8; ++j) Bs[buf][n][kk0 + j] = (__bf16)bRegF[j];
    } else if constexpr (!BT) {
      const int n = tid & 63;
      const int kk0 = (tid >> 6) << 3;
#pragma unroll
      for (int j = 0; j < 8; ++j) Bs[buf][n][kk0 + j] = bRegH[j];
    }
  };

  v8f acc[2][2] = {};

  // prologue: tile 0 -> buffer 0
  issueA(0, 0);
  issueB(0, 0);
  commitA(0);
  commitB(0);
  async_wait0();
  __syncthreads();

  const int nk = K >> 5;
  for (int kt = 0; kt < nk; ++kt) {
    const int cur = kt & 1;
    const bool more = (kt + 1 < nk);
    if (more) {  // start next tile before computing on current one
      issueA((kt + 1) << 5, cur ^ 1);
      issueB((kt + 1) << 5, cur ^ 1);
    }

    // fragments per ISA 7.12.2 layouts
    v16bf af[2], bfv[2];
#pragma unroll
    for (int i = 0; i < 2; ++i) {
      const __bf16* pr = &As[cur][wm * 32 + i * 16 + mi][0];
      v8bf lo = *(const v8bf*)(pr + kg * 8);
      v8bf hi = *(const v8bf*)(pr + kg * 8 + 16);
#pragma unroll
      for (int e = 0; e < 8; ++e) {
        af[i][e] = lo[e];
        af[i][8 + e] = hi[e];
      }
    }
#pragma unroll
    for (int j = 0; j < 2; ++j) {
      const __bf16* pr = &Bs[cur][wn * 32 + j * 16 + mi][0];
      v8bf lo = *(const v8bf*)(pr + kg * 8);
      v8bf hi = *(const v8bf*)(pr + kg * 8 + 16);
#pragma unroll
      for (int e = 0; e < 8; ++e) {
        bfv[j][e] = lo[e];
        bfv[j][8 + e] = hi[e];
      }
    }
#pragma unroll
    for (int i = 0; i < 2; ++i)
#pragma unroll
      for (int j = 0; j < 2; ++j)
        acc[i][j] = __builtin_amdgcn_wmma_f32_16x16x32_bf16(
            false, af[i], false, bfv[j], (short)0, acc[i][j], false, false);

    if (more) {  // convert/commit after the WMMAs have been issued
      commitA(cur ^ 1);
      commitB(cur ^ 1);
    }
    async_wait0();
    __syncthreads();
  }

  // epilogue: C layout row = kg*8+v, col = lane&15
#pragma unroll
  for (int i = 0; i < 2; ++i)
#pragma unroll
    for (int j = 0; j < 2; ++j) {
      const int col = tN + wn * 32 + j * 16 + mi;
#pragma unroll
      for (int v = 0; v < 8; ++v) {
        const int row = tM + wm * 32 + i * 16 + kg * 8 + v;
        if (row < M && col < N) {
          float val = alpha * acc[i][j][v];
          if constexpr (EPI >= EPI_BIAS) val += bias[col];
          if constexpr (EPI == EPI_RELU) val = fmaxf(val, 0.f);
          if constexpr (EPI == EPI_GELU)
            val = 0.5f * val * (1.f + erff(val * 0.70710678118654752f));
          if constexpr (EPI == EPI_RES) val += (float)R[(size_t)row * ldr + col];
          C[(size_t)row * ldc + col] = (TC)val;
        }
      }
    }
}

// ---------------------------------------------------------------------------
// Elementwise kernels (bf16 activations, fp32 math)
// ---------------------------------------------------------------------------
__global__ __launch_bounds__(128) void ln384_kernel(
    const __bf16* __restrict__ x, const float* __restrict__ g,
    const float* __restrict__ b, __bf16* __restrict__ out) {
  const int row = blockIdx.x;
  const __bf16* xr = x + (size_t)row * 384;
  const int t = threadIdx.x;
  const float v0 = (float)xr[t], v1 = (float)xr[t + 128], v2 = (float)xr[t + 256];
  __shared__ float sm[4];
  float s = wave_sum(v0 + v1 + v2);
  if ((t & 31) == 0) sm[t >> 5] = s;
  __syncthreads();
  const float mean = (sm[0] + sm[1] + sm[2] + sm[3]) * (1.f / 384.f);
  const float d0 = v0 - mean, d1 = v1 - mean, d2 = v2 - mean;
  float q = wave_sum(d0 * d0 + d1 * d1 + d2 * d2);
  __syncthreads();
  if ((t & 31) == 0) sm[t >> 5] = q;
  __syncthreads();
  const float inv = rsqrtf((sm[0] + sm[1] + sm[2] + sm[3]) * (1.f / 384.f) + 1e-5f);
  __bf16* o = out + (size_t)row * 384;
  o[t] = (__bf16)(d0 * inv * g[t] + b[t]);
  o[t + 128] = (__bf16)(d1 * inv * g[t + 128] + b[t + 128]);
  o[t + 256] = (__bf16)(d2 * inv * g[t + 256] + b[t + 256]);
}

__global__ __launch_bounds__(128) void softmax512_kernel(__bf16* __restrict__ p_) {
  __bf16* p = p_ + (size_t)blockIdx.x * 512;
  const int t = threadIdx.x;
  float v[4];
#pragma unroll
  for (int i = 0; i < 4; ++i) v[i] = (float)p[t + i * 128];
  __shared__ float sm[4];
  float m = wave_max(fmaxf(fmaxf(v[0], v[1]), fmaxf(v[2], v[3])));
  if ((t & 31) == 0) sm[t >> 5] = m;
  __syncthreads();
  m = fmaxf(fmaxf(sm[0], sm[1]), fmaxf(sm[2], sm[3]));
  float s = 0.f;
#pragma unroll
  for (int i = 0; i < 4; ++i) {
    v[i] = __expf(v[i] - m);
    s += v[i];
  }
  s = wave_sum(s);
  __syncthreads();
  if ((t & 31) == 0) sm[t >> 5] = s;
  __syncthreads();
  const float inv = 1.f / (sm[0] + sm[1] + sm[2] + sm[3]);
#pragma unroll
  for (int i = 0; i < 4; ++i) p[t + i * 128] = (__bf16)(v[i] * inv);
}

__global__ __launch_bounds__(128) void mask_kernel(const float* __restrict__ gf,
                                                   float* __restrict__ mask) {
  const int row = blockIdx.x;
  const float* r = gf + (size_t)row * 512;
  const int t = threadIdx.x;
  float s = 0.f;
#pragma unroll
  for (int i = 0; i < 4; ++i) s += fabsf(r[t + i * 128]);
  s = wave_sum(s);
  __shared__ float sm[4];
  if ((t & 31) == 0) sm[t >> 5] = s;
  __syncthreads();
  if (t == 0) mask[row] = ((sm[0] + sm[1] + sm[2] + sm[3]) == 0.f) ? 1.f : 0.f;
}

// scores += att_bias[h]; zero masked v-rows.  scores: [B,4,512,512] fp32
__global__ __launch_bounds__(256) void scores_fix_kernel(
    float* __restrict__ s, const float* __restrict__ att_bias,
    const float* __restrict__ mask) {
  const long long i = (long long)blockIdx.x * 256 + threadIdx.x;
  const int b = (int)(i >> 20);
  const int h = (int)((i >> 18) & 3);
  const int v = (int)((i >> 9) & 511);
  s[i] = (mask[b * 512 + v] != 0.f) ? 0.f : (s[i] + att_bias[h]);
}

// Ssum[b,v,q] = sum_h scores[b,h,v,q]  (fp32 -> bf16 for the next GEMM's A)
__global__ __launch_bounds__(256) void headsum_kernel(
    const float* __restrict__ s, __bf16* __restrict__ out) {
  const long long i = (long long)blockIdx.x * 256 + threadIdx.x;  // < 2^23
  const int b = (int)(i >> 18);
  const int r = (int)(i & 262143);
  const float* p = s + ((long long)b << 20) + r;
  out[i] = (__bf16)(p[0] + p[262144] + p[524288] + p[786432]);
}

__global__ __launch_bounds__(384) void final_kernel(
    const __bf16* __restrict__ ga, const float* __restrict__ t_,
    const float* __restrict__ bn_g, const float* __restrict__ bn_b,
    const float* __restrict__ bn_mean, const float* __restrict__ bn_var,
    const float* __restrict__ Wf, const float* __restrict__ bfc,
    float* __restrict__ out) {
  const int b = blockIdx.x;
  const int d = threadIdx.x;  // 0..383
  const __bf16* g = ga + (size_t)b * 196608;
  const float* t = t_ + (size_t)b * 196608;
  float acc = 0.f;
  for (int v = 0; v < 512; ++v)
    acc += (float)g[(size_t)v * 384 + d] * t[(size_t)v * 384 + d];
  __shared__ float logits[384];
  __shared__ float contrib[128];
  logits[d] = acc;
  __syncthreads();
  if (d < 128) {
    const float pooled = logits[3 * d] + logits[3 * d + 1] + logits[3 * d + 2];
    const float bn =
        (pooled - bn_mean[d]) * rsqrtf(bn_var[d] + 1e-5f) * bn_g[d] + bn_b[d];
    contrib[d] = bn * Wf[d];
  }
  __syncthreads();
  if (d == 0) {
    float s = bfc[0];
    for (int i = 0; i < 128; ++i) s += contrib[i];
    out[b] = s;
  }
}

// ---------------------------------------------------------------------------
// Host side
// ---------------------------------------------------------------------------
template <int EPI, bool BT, typename TA, typename TB, typename TC>
static void gemm(hipStream_t st, const TA* A, const TB* B, TC* C,
                 const float* bias, const TC* R, int M, int N, int K, int lda,
                 int ldb, int ldc, int ldr, long long sAb, long long sAh,
                 long long sBb, long long sBh, long long sCb, long long sCh,
                 long long sRb, long long sRh, int Hdim, int batches,
                 float alpha) {
  dim3 grid((N + TN - 1) / TN, (M + TM - 1) / TM, batches);
  gemm_wmma_k<EPI, BT, TA, TB, TC><<<grid, 256, 0, st>>>(
      A, B, C, bias, R, M, N, K, lda, ldb, ldc, ldr, sAb, sAh, sBb, sBh, sCb,
      sCh, sRb, sRh, Hdim, alpha);
}

static void run_transformer(hipStream_t st, __bf16* x, __bf16* h, __bf16* qkv,
                            __bf16* dots, __bf16* ao, __bf16* ff,
                            const float* ln1_g, const float* ln1_b,
                            const float* Wqkv, const float* Wo, const float* bo,
                            const float* ln2_g, const float* ln2_b,
                            const float* W1, const float* b1, const float* W2,
                            const float* b2) {
  const int M = 32 * 512;
  ln384_kernel<<<M, 128, 0, st>>>(x, ln1_g, ln1_b, h);
  gemm<EPI_NONE, false, __bf16, float, __bf16>(
      st, h, Wqkv, qkv, nullptr, nullptr, M, 288, 384, 384, 288, 288, 0, 0, 0,
      0, 0, 0, 0, 0, 0, 1, 1, 1.f);
  gemm<EPI_NONE, true, __bf16, __bf16, __bf16>(
      st, qkv, qkv + 96, dots, nullptr, nullptr, 512, 512, 32, 288, 288, 512,
      0, 147456, 32, 147456, 32, 786432, 262144, 0, 0, 3, 96,
      0.17677669529663687f);
  softmax512_kernel<<<96 * 512, 128, 0, st>>>(dots);
  gemm<EPI_NONE, false, __bf16, __bf16, __bf16>(
      st, dots, qkv + 192, ao, nullptr, nullptr, 512, 32, 512, 512, 288, 96, 0,
      786432, 262144, 147456, 32, 49152, 32, 0, 0, 3, 96, 1.f);
  gemm<EPI_RES, false, __bf16, float, __bf16>(
      st, ao, Wo, x, bo, x, M, 384, 96, 96, 384, 384, 384, 0, 0, 0, 0, 0, 0, 0,
      0, 1, 1, 1.f);
  ln384_kernel<<<M, 128, 0, st>>>(x, ln2_g, ln2_b, h);
  gemm<EPI_GELU, false, __bf16, float, __bf16>(
      st, h, W1, ff, b1, nullptr, M, 128, 384, 384, 128, 128, 0, 0, 0, 0, 0, 0,
      0, 0, 0, 1, 1, 1.f);
  gemm<EPI_RES, false, __bf16, float, __bf16>(
      st, ff, W2, x, b2, x, M, 384, 128, 128, 384, 384, 384, 0, 0, 0, 0, 0, 0,
      0, 0, 1, 1, 1.f);
}

extern "C" void kernel_launch(void* const* d_in, const int* in_sizes, int n_in,
                              void* d_out_, int out_size, void* d_ws,
                              size_t ws_size, hipStream_t stream) {
  (void)in_sizes; (void)n_in; (void)out_size; (void)ws_size;
  const float* global_feat = (const float*)d_in[0];
  const float* local_feat = (const float*)d_in[1];
  const float* Wg = (const float*)d_in[2];
  const float* bg = (const float*)d_in[3];
  const float* Wl = (const float*)d_in[4];
  const float* bl = (const float*)d_in[5];
  const float* ln1_g = (const float*)d_in[6];
  const float* ln1_b = (const float*)d_in[7];
  const float* Wqkv = (const float*)d_in[8];
  const float* Wo = (const float*)d_in[9];
  const float* bo = (const float*)d_in[10];
  const float* ln2_g = (const float*)d_in[11];
  const float* ln2_b = (const float*)d_in[12];
  const float* W1 = (const float*)d_in[13];
  const float* b1 = (const float*)d_in[14];
  const float* W2 = (const float*)d_in[15];
  const float* b2 = (const float*)d_in[16];
  const float* Wq = (const float*)d_in[17];
  const float* Wk = (const float*)d_in[18];
  const float* att_bias = (const float*)d_in[19];
  const float* bn_g = (const float*)d_in[20];
  const float* bn_b = (const float*)d_in[21];
  const float* bn_mean = (const float*)d_in[22];
  const float* bn_var = (const float*)d_in[23];
  const float* Wf = (const float*)d_in[24];
  const float* bfc = (const float*)d_in[25];

  float* out = (float*)d_out_;
  float* scores = out + 32;  // [32,4,512,512] fp32 (part of the output tuple)

  char* wsp = (char*)d_ws;
  size_t off = 0;
  auto alloc = [&](size_t bytes) -> void* {
    void* p = wsp + off;
    off += (bytes + 255) & ~(size_t)255;
    return p;
  };
  __bf16* GA = (__bf16*)alloc((size_t)16384 * 384 * 2);
  __bf16* LA = (__bf16*)alloc((size_t)16384 * 384 * 2);
  __bf16* H = (__bf16*)alloc((size_t)16384 * 384 * 2);
  __bf16* QKV = (__bf16*)alloc((size_t)16384 * 288 * 2);
  __bf16* AO = (__bf16*)alloc((size_t)16384 * 96 * 2);
  __bf16* FF = (__bf16*)alloc((size_t)16384 * 128 * 2);
  __bf16* Qc = (__bf16*)alloc((size_t)16384 * 384 * 2);
  __bf16* Kc = (__bf16*)alloc((size_t)16384 * 384 * 2);
  __bf16* DOTS = (__bf16*)alloc((size_t)96 * 512 * 512 * 2);  // also Ssum
  float* T = (float*)alloc((size_t)32 * 512 * 384 * 4);
  float* MASK = (float*)alloc((size_t)16384 * 4);

  const int M = 32 * 512;

  // Input projections: fp32 in, fused bias+ReLU, bf16 out
  gemm<EPI_RELU, false, float, float, __bf16>(
      stream, global_feat, Wg, GA, bg, nullptr, M, 384, 512, 512, 384, 384, 0,
      0, 0, 0, 0, 0, 0, 0, 0, 1, 1, 1.f);
  gemm<EPI_RELU, false, float, float, __bf16>(
      stream, local_feat, Wl, LA, bl, nullptr, M, 384, 256, 256, 384, 384, 0,
      0, 0, 0, 0, 0, 0, 0, 0, 1, 1, 1.f);

  run_transformer(stream, GA, H, QKV, DOTS, AO, FF, ln1_g, ln1_b, Wqkv, Wo, bo,
                  ln2_g, ln2_b, W1, b1, W2, b2);
  run_transformer(stream, LA, H, QKV, DOTS, AO, FF, ln1_g, ln1_b, Wqkv, Wo, bo,
                  ln2_g, ln2_b, W1, b1, W2, b2);

  // Cross projections
  gemm<EPI_NONE, false, __bf16, float, __bf16>(
      stream, GA, Wq, Qc, nullptr, nullptr, M, 384, 384, 384, 384, 384, 0, 0,
      0, 0, 0, 0, 0, 0, 0, 1, 1, 1.f);
  gemm<EPI_NONE, false, __bf16, float, __bf16>(
      stream, LA, Wk, Kc, nullptr, nullptr, M, 384, 384, 384, 384, 384, 0, 0,
      0, 0, 0, 0, 0, 0, 0, 1, 1, 1.f);

  mask_kernel<<<M, 128, 0, stream>>>(global_feat, MASK);

  // scores[b,h] = q @ k^T / sqrt(96) -> fp32, straight into d_out
  gemm<EPI_NONE, true, __bf16, __bf16, float>(
      stream, Qc, Kc, scores, nullptr, nullptr, 512, 512, 96, 384, 384, 512, 0,
      196608, 96, 196608, 96, 1048576, 262144, 0, 0, 4, 128,
      0.10206207261596575f);
  scores_fix_kernel<<<131072, 256, 0, stream>>>(scores, att_bias, MASK);

  // Head-sum (ga/la are h-independent in the triple contraction)
  headsum_kernel<<<32768, 256, 0, stream>>>(scores, DOTS);

  // T[b] = Ssum[b] @ la[b]  (512x384x512, 32 batches), fp32 out
  gemm<EPI_NONE, false, __bf16, __bf16, float>(
      stream, DOTS, LA, T, nullptr, nullptr, 512, 384, 512, 512, 384, 384, 0,
      262144, 0, 196608, 0, 196608, 0, 0, 0, 1, 32, 1.f);

  // logits = sum_v ga*T; pool3 -> BN -> @Wf + bf
  final_kernel<<<32, 384, 0, stream>>>(GA, T, bn_g, bn_b, bn_mean, bn_var, Wf,
                                       bfc, out);
}
